// EPA_attn_41738492183085
// MI455X (gfx1250) — compile-verified, matrix operand order checked by reading, not run on hardware
//
#include <hip/hip_runtime.h>
#include <hip/hip_bf16.h>

// ---------------------------------------------------------------------------
// EPA attention for MI455X (gfx1250, wave32). All contractions via
// v_wmma_f32_16x16x32_bf16 (fp32 storage, bf16 convert-on-load, f32 accum).
// The fused Linformer-attention kernel stages its projection matrices into
// LDS with global_load_async_to_lds_b128 (ASYNCcnt) before the WMMA loop.
// ---------------------------------------------------------------------------

typedef __attribute__((ext_vector_type(16))) __bf16 v16bf;
typedef __attribute__((ext_vector_type(8)))  float  v8f;

#define NTOK 32768     // H*W*D
#define BATCH 2
#define NBH 8          // B*h
#define DH 32
#define PDIM 64

// ---- workspace layout (in floats) ----
constexpr size_t SZ_BCN = (size_t)BATCH * 64 * NTOK;      // 4,194,304
constexpr size_t SZ_Q   = (size_t)NBH * DH * NTOK;        // 8,388,608
constexpr size_t O_X1F  = 0;
constexpr size_t O_X2F  = O_X1F + SZ_BCN;
constexpr size_t O_XCAT = O_X2F + SZ_BCN;                 // [B*N][128]
constexpr size_t O_Q    = O_XCAT + (size_t)BATCH * NTOK * 128;
constexpr size_t O_K    = O_Q   + SZ_Q;
constexpr size_t O_VCA  = O_K   + SZ_Q;
constexpr size_t O_VSA  = O_VCA + SZ_Q;
constexpr size_t O_XCA  = O_VSA + SZ_Q;
constexpr size_t O_XSA  = O_XCA + SZ_Q;
constexpr size_t O_KPROJ= O_XSA + SZ_Q;                   // [NBH][32][64]
constexpr size_t O_VPROJ= O_KPROJ + (size_t)NBH * DH * PDIM;
constexpr size_t O_QINV = O_VPROJ + (size_t)NBH * DH * PDIM;
constexpr size_t O_KINV = O_QINV + NBH * DH;
constexpr size_t O_SCA  = O_KINV + NBH * DH;              // [NBH][32][32]

// ---- WMMA helpers -----------------------------------------------------------
// A 16x32 bf16 fragment (ISA 7.12.2): lane = M row (lane&15), lanes 16-31 get
// K+8; VGPR r<4 -> K=2r, r>=4 -> K=16+2(r-4); dword packs (K,K+1).
__device__ __forceinline__ int kpos(int e, int half) {
  int r = e >> 1;
  return ((r & 3) * 2 + (e & 1)) + ((r & 4) << 2) + half * 8;
}

__device__ __forceinline__ v8f vzero() {
  v8f z;
#pragma unroll
  for (int i = 0; i < 8; ++i) z[i] = 0.0f;
  return z;
}

__device__ __forceinline__ v8f wmma_bf16(v16bf a, v16bf b, v8f c) {
  return __builtin_amdgcn_wmma_f32_16x16x32_bf16(false, a, false, b,
                                                 (short)0, c, false, false);
}

// source laid out [row][k] row-major, leading dim ld (row = M for A, N for Bt)
__device__ __forceinline__ v16bf frag_rowmajor(const float* __restrict__ p,
                                               int ld, int lane) {
  int m = lane & 15, half = lane >> 4;
  v16bf f;
#pragma unroll
  for (int e = 0; e < 16; ++e)
    f[e] = (__bf16)p[(size_t)m * ld + kpos(e, half)];
  return f;
}

// source laid out [k][row] (K-strided), leading dim ld; lane-coalesced
__device__ __forceinline__ v16bf frag_kstrided(const float* __restrict__ p,
                                               int ld, int lane) {
  int m = lane & 15, half = lane >> 4;
  v16bf f;
#pragma unroll
  for (int e = 0; e < 16; ++e)
    f[e] = (__bf16)p[(size_t)kpos(e, half) * ld + m];
  return f;
}

// C/D 16x16 f32: lane col = lane&15, VGPR v holds M = v + 8*(lane>>4)
__device__ __forceinline__ void store_tile(float* __restrict__ p, int ld,
                                           int lane, v8f c) {
  int col = lane & 15, half = lane >> 4;
#pragma unroll
  for (int v = 0; v < 8; ++v)
    p[(size_t)(v + 8 * half) * ld + col] = c[v];
}

__device__ __forceinline__ void atomic_tile(float* __restrict__ p, int ld,
                                            int lane, v8f c) {
  int col = lane & 15, half = lane >> 4;
#pragma unroll
  for (int v = 0; v < 8; ++v)
    unsafeAtomicAdd(&p[(size_t)(v + 8 * half) * ld + col], c[v]);
}

// async copy of one 16B granule global -> LDS (ASYNCcnt-tracked)
__device__ __forceinline__ void async_b128_to_lds(const float* g, float* l) {
  unsigned lds  = (unsigned)(unsigned long long)l;
  unsigned long long ga = (unsigned long long)g;
  asm volatile("global_load_async_to_lds_b128 %0, %1, off"
               :: "v"(lds), "v"(ga) : "memory");
}

__device__ __forceinline__ void wait_async0() {
  asm volatile("s_wait_asynccnt 0x0" ::: "memory");
}

// ---------------------------------------------------------------------------
// 1) LayerNorm over C_=64 per token; writes [B][64][N] (residual layout) and
//    token-major concat buffer [B*N][128] (GEMM A). Loads coalesced over n.
__global__ void k_layernorm(const float* __restrict__ src,
                            const float* __restrict__ w,
                            const float* __restrict__ bvec,
                            float* __restrict__ xf,
                            float* __restrict__ xcat) {
  int t = blockIdx.x * blockDim.x + threadIdx.x;   // 0..B*N-1
  int b = t >> 15, n = t & (NTOK - 1);
  const float* s = src + (size_t)b * 64 * NTOK + n;
  float v[64];
  float mu = 0.f;
#pragma unroll
  for (int c = 0; c < 64; ++c) { v[c] = s[(size_t)c * NTOK]; mu += v[c]; }
  mu *= (1.f / 64.f);
  float var = 0.f;
#pragma unroll
  for (int c = 0; c < 64; ++c) { float d = v[c] - mu; var += d * d; }
  var *= (1.f / 64.f);
  float r = rsqrtf(var + 1e-5f);
  float* xo = xf + (size_t)b * 64 * NTOK + n;
  float* xc = xcat + (size_t)t * 128;
#pragma unroll
  for (int c = 0; c < 64; ++c) {
    float y = (v[c] - mu) * r * w[c] + bvec[c];
    xo[(size_t)c * NTOK] = y;
    xc[c] = y;
  }
}

// 2) init: kproj = E_b, vproj = F_b (split-K atomics accumulate on top),
//    S_ca = 0
__global__ void k_init(const float* __restrict__ Eb, const float* __restrict__ Fb,
                       float* __restrict__ kproj, float* __restrict__ vproj,
                       float* __restrict__ sca) {
  int i = blockIdx.x * blockDim.x + threadIdx.x;
  if (i < NBH * DH * PDIM) {
    kproj[i] = Eb[i & 63];
    vproj[i] = Fb[i & 63];
  }
  if (i < NBH * DH * DH) sca[i] = 0.f;
}

// 3) qkvv GEMM: out[d,t] = sum_k W[d,k] * xcat[t,k]; D rows = d (const
//    which/head per 16-tile), cols = tokens -> coalesced stores into
//    q/k/vca/vsa [bh][c][N].
__global__ void __launch_bounds__(256)
k_gemm_qkvv(const float* __restrict__ xcat, const float* __restrict__ W,
            float* __restrict__ q, float* __restrict__ k,
            float* __restrict__ vca, float* __restrict__ vsa) {
  int id = blockIdx.x * 8 + (threadIdx.x >> 5);
  int lane = threadIdx.x & 31;
  int tn = id & 4095;            // token tile
  int tm = id >> 12;             // d tile (0..31)
  int d0 = tm * 16, t0 = tn * 16;
  v8f acc = vzero();
#pragma unroll
  for (int kk = 0; kk < 128; kk += 32) {
    v16bf a = frag_rowmajor(W + (size_t)d0 * 128 + kk, 128, lane);
    v16bf b = frag_rowmajor(xcat + (size_t)t0 * 128 + kk, 128, lane);
    acc = wmma_bf16(a, b, acc);
  }
  int bb = t0 >> 15, n0 = t0 & (NTOK - 1);
  int which = d0 >> 7, hh = (d0 >> 5) & 3;
  float* arr = which == 0 ? q : which == 1 ? k : which == 2 ? vca : vsa;
  int col = lane & 15, half = lane >> 4;
  float* basep = arr + ((size_t)((bb * 4 + hh) * DH + (d0 & 31))) * NTOK + n0 + col;
#pragma unroll
  for (int v = 0; v < 8; ++v)
    basep[(size_t)(v + 8 * half) * NTOK] = acc[v];
}

// 4) row sum-of-squares over N=32768 -> 1/max(||.||, 1e-12) for q and k
__global__ void k_rowsumsq(const float* __restrict__ q,
                           const float* __restrict__ k,
                           float* __restrict__ qinv, float* __restrict__ kinv) {
  int row = blockIdx.x;                      // 0..511 (first 256 q, then k)
  const float* src = (row < 256 ? q : k) + (size_t)(row & 255) * NTOK;
  float s = 0.f;
  for (int i = threadIdx.x; i < NTOK; i += 256) { float x = src[i]; s += x * x; }
  __shared__ float red[256];
  red[threadIdx.x] = s;
  __syncthreads();
  for (int off = 128; off; off >>= 1) {
    if (threadIdx.x < off) red[threadIdx.x] += red[threadIdx.x + off];
    __syncthreads();
  }
  if (threadIdx.x == 0) {
    float inv = 1.f / fmaxf(sqrtf(red[0]), 1e-12f);
    (row < 256 ? qinv : kinv)[row & 255] = inv;
  }
}

// 5) generic split-K GEMM over K=32768: out[bh][m][n] (+=) A[bh-rows][N] x
//    Bt[rows][N]. Used for k_proj, v_proj, and raw CA logits. Partial sums
//    accumulated with global f32 atomics.
__global__ void __launch_bounds__(256)
k_gemm_splitk(const float* __restrict__ A, long a_stride,
              const float* __restrict__ Bt, long b_stride,
              float* __restrict__ out, long o_stride, int o_ld,
              int mt_cnt, int nt_cnt, int ksplit, int kchunk) {
  int id = blockIdx.x * 8 + (threadIdx.x >> 5);
  int lane = threadIdx.x & 31;
  int ks = id % ksplit; id /= ksplit;
  int tn = id % nt_cnt; id /= nt_cnt;
  int tm = id % mt_cnt;
  int bh = id / mt_cnt;
  const float* a  = A  + (size_t)bh * a_stride + (size_t)(tm * 16) * NTOK + (size_t)ks * kchunk;
  const float* bt = Bt + (size_t)bh * b_stride + (size_t)(tn * 16) * NTOK + (size_t)ks * kchunk;
  v8f acc = vzero();
  for (int kk = 0; kk < kchunk; kk += 32) {
    __builtin_prefetch(a + kk + 256, 0, 3);
    __builtin_prefetch(bt + kk + 256, 0, 3);
    v16bf af = frag_rowmajor(a + kk, NTOK, lane);
    v16bf bf = frag_rowmajor(bt + kk, NTOK, lane);
    acc = wmma_bf16(af, bf, acc);
  }
  atomic_tile(out + (size_t)bh * o_stride + (size_t)(tm * 16) * o_ld + tn * 16,
              o_ld, lane, acc);
}

// 6) CA softmax over d (32) with l2-norm scaling and temperature; one wave/row
__global__ void k_softmax_ca(float* __restrict__ sca,
                             const float* __restrict__ qinv,
                             const float* __restrict__ kinv,
                             const float* __restrict__ temp) {
  int row = blockIdx.x;           // bh*32 + c
  int bh = row >> 5, hh = bh & 3;
  int d = threadIdx.x;            // 0..31
  float v = sca[row * 32 + d] * qinv[row] * kinv[bh * 32 + d] * temp[hh];
  float m = v;
  for (int off = 16; off; off >>= 1) m = fmaxf(m, __shfl_xor(m, off));
  float e = __expf(v - m);
  float s = e;
  for (int off = 16; off; off >>= 1) s += __shfl_xor(s, off);
  sca[row * 32 + d] = e / s;
}

// 7) x_ca[bh][c][n] = probs[bh][c][d] @ vca[bh][d][n], K=32 (one WMMA)
__global__ void __launch_bounds__(256)
k_gemm_xca(const float* __restrict__ sca, const float* __restrict__ vca,
           float* __restrict__ xca) {
  int id = blockIdx.x * 8 + (threadIdx.x >> 5);
  int lane = threadIdx.x & 31;
  int tn = id & 2047, tm = (id >> 11) & 1, bh = id >> 12;
  v16bf a = frag_rowmajor(sca + (size_t)bh * (DH * DH) + (size_t)tm * 16 * DH, DH, lane);
  v16bf b = frag_kstrided(vca + (size_t)bh * DH * NTOK + tn * 16, NTOK, lane);
  v8f acc = wmma_bf16(a, b, vzero());
  store_tile(xca + ((size_t)bh * DH + tm * 16) * NTOK + tn * 16, NTOK, lane, acc);
}

// 8) Fused spatial (Linformer) attention per 16-token tile:
//    async-stage kproj/vproj[bh] (16 KB) into LDS (all 8 waves of a block
//    share the same bh) -> logits (WMMA, K=32) -> softmax in LDS ->
//    probs @ vproj^T (WMMA, K=64)
__global__ void __launch_bounds__(256)
k_sa_fused(const float* __restrict__ q, const float* __restrict__ qinv,
           const float* __restrict__ kproj, const float* __restrict__ vproj,
           const float* __restrict__ t2v, float* __restrict__ xsa) {
  __shared__ float L[8][16][64];            // per-wave logits/probs tile
  __shared__ float KP[DH * PDIM];           // kproj[bh]  (2048 f)
  __shared__ float VP[DH * PDIM];           // vproj[bh]  (2048 f)
  int wv = threadIdx.x >> 5, lane = threadIdx.x & 31;
  int id = blockIdx.x * 8 + wv;
  int tile = id & 2047, bh = id >> 11;      // bh uniform across the block
  int n0 = tile * 16;
  int m = lane & 15, half = lane >> 4;
  float (*Lw)[64] = L[wv];

  // ---- async global->LDS staging of the projection matrices ----
  {
    const float* gk = kproj + (size_t)bh * (DH * PDIM);
    const float* gv = vproj + (size_t)bh * (DH * PDIM);
    int tid = threadIdx.x;
#pragma unroll
    for (int r = 0; r < 2; ++r) {           // 2 x 256 threads x 16B = 8 KB
      int i = (r * 256 + tid) * 4;
      async_b128_to_lds(gk + i, &KP[i]);
      async_b128_to_lds(gv + i, &VP[i]);
    }
    wait_async0();
    __syncthreads();
  }

  // A[m=token][k=c] = qn, gathered from q[bh][c][n] (lane-coalesced in n)
  const float* qb = q + (size_t)bh * DH * NTOK + n0;
  v16bf a;
#pragma unroll
  for (int e = 0; e < 16; ++e) {
    int kc = kpos(e, half);
    a[e] = (__bf16)(qb[(size_t)kc * NTOK + m] * qinv[bh * DH + kc]);
  }
  float t2 = t2v[bh & 3];
#pragma unroll
  for (int pt = 0; pt < 4; ++pt) {
    // Bt fragment from LDS: KP laid out [c][p], K-strided, ld = PDIM
    v16bf b;
#pragma unroll
    for (int e = 0; e < 16; ++e)
      b[e] = (__bf16)KP[kpos(e, half) * PDIM + pt * 16 + m];
    v8f acc = wmma_bf16(a, b, vzero());
#pragma unroll
    for (int v = 0; v < 8; ++v)
      Lw[v + 8 * half][pt * 16 + m] = acc[v] * t2;
  }
  __syncthreads();
  if (lane < 16) {                 // lanes 0..15 each own one token row
    float mx = -1e30f;
#pragma unroll
    for (int j = 0; j < 64; ++j) mx = fmaxf(mx, Lw[lane][j]);
    float s = 0.f;
#pragma unroll
    for (int j = 0; j < 64; ++j) {
      float e = __expf(Lw[lane][j] - mx);
      Lw[lane][j] = e;
      s += e;
    }
    float inv = 1.f / s;
#pragma unroll
    for (int j = 0; j < 64; ++j) Lw[lane][j] *= inv;
  }
  __syncthreads();
#pragma unroll
  for (int ct = 0; ct < 2; ++ct) {  // output channels c in two 16-tiles
    v8f o = vzero();
#pragma unroll
    for (int kk = 0; kk < PDIM; kk += 32) {
      v16bf a2;
#pragma unroll
      for (int e = 0; e < 16; ++e)
        a2[e] = (__bf16)Lw[m][kk + kpos(e, half)];
      // Bt fragment from LDS: VP laid out [c][p] row-major, ld = PDIM
      v16bf b2;
#pragma unroll
      for (int e = 0; e < 16; ++e)
        b2[e] = (__bf16)VP[(ct * 16 + m) * PDIM + kk + kpos(e, half)];
      o = wmma_bf16(a2, b2, o);
    }
    store_tile(xsa + ((size_t)bh * NTOK + n0) * DH + ct * 16, DH, lane, o);
  }
}

// 9) out1[b][ch][n] = x1f + gamma*(xsa_scrambled @ W^T + b). The reference's
//    permute(0,3,1,2).reshape gather is inverted analytically:
//    flat = t'*128+ch; u = flat>>15 (lane-constant); c=u>>2, hh=u&3,
//    n = ((t'&255)<<7)|ch  ->  per-lane base + ch*32 strided reads.
__global__ void __launch_bounds__(256)
k_outproj_sa(const float* __restrict__ xsa, const float* __restrict__ Wo,
             const float* __restrict__ bo, const float* __restrict__ gamma,
             const float* __restrict__ x1f, float* __restrict__ outp) {
  int id = blockIdx.x * 8 + (threadIdx.x >> 5);
  int lane = threadIdx.x & 31;
  int tn = id & 4095, tm = id >> 12;        // tm 0..3 (channel tiles)
  int ch0 = tm * 16, t0 = tn * 16;
  int bb = t0 >> 15;
  int m = lane & 15, half = lane >> 4;
  int tprime = (t0 & (NTOK - 1)) + m;       // token within batch (B col)
  int u = tprime >> 8;
  int cc = u >> 2, hh = u & 3;
  int nn = (tprime & 255) << 7;
  const float* bbase = xsa + ((size_t)(bb * 4 + hh) * NTOK + nn) * DH + cc;
  v8f acc = vzero();
#pragma unroll
  for (int kk = 0; kk < 128; kk += 32) {
    v16bf a = frag_rowmajor(Wo + (size_t)ch0 * 128 + kk, 128, lane);
    v16bf b;
#pragma unroll
    for (int e = 0; e < 16; ++e)
      b[e] = (__bf16)bbase[(size_t)(kk + kpos(e, half)) * DH];
    acc = wmma_bf16(a, b, acc);
  }
  size_t ob = (size_t)bb * 64 * NTOK + (t0 & (NTOK - 1)) + m;
#pragma unroll
  for (int v = 0; v < 8; ++v) {
    int ch = ch0 + v + 8 * half;
    size_t idx = ob + (size_t)ch * NTOK;
    outp[idx] = x1f[idx] + gamma[ch] * (acc[v] + bo[ch]);
  }
}

// 10) out2[b][ch][n] = x2f + gamma*(xca_tokens @ W2^T + b2); x_ca reshape is
//     natural: channel k -> row (b*128+k) of [B][128][N].
__global__ void __launch_bounds__(256)
k_outproj_ca(const float* __restrict__ xca, const float* __restrict__ Wo,
             const float* __restrict__ bo, const float* __restrict__ gamma,
             const float* __restrict__ x2f, float* __restrict__ outp) {
  int id = blockIdx.x * 8 + (threadIdx.x >> 5);
  int lane = threadIdx.x & 31;
  int tn = id & 4095, tm = id >> 12;
  int ch0 = tm * 16, t0 = tn * 16;
  int bb = t0 >> 15;
  int m = lane & 15, half = lane >> 4;
  int nloc = (t0 & (NTOK - 1)) + m;
  const float* bbase = xca + (size_t)bb * 128 * NTOK + nloc;
  v8f acc = vzero();
#pragma unroll
  for (int kk = 0; kk < 128; kk += 32) {
    v16bf a = frag_rowmajor(Wo + (size_t)ch0 * 128 + kk, 128, lane);
    v16bf b;
#pragma unroll
    for (int e = 0; e < 16; ++e)
      b[e] = (__bf16)bbase[(size_t)(kk + kpos(e, half)) * NTOK];
    acc = wmma_bf16(a, b, acc);
  }
  size_t ob = (size_t)bb * 64 * NTOK + (t0 & (NTOK - 1)) + m;
#pragma unroll
  for (int v = 0; v < 8; ++v) {
    int ch = ch0 + v + 8 * half;
    size_t idx = ob + (size_t)ch * NTOK;
    outp[idx] = x2f[idx] + gamma[ch] * (acc[v] + bo[ch]);
  }
}

// ---------------------------------------------------------------------------
extern "C" void kernel_launch(void* const* d_in, const int* in_sizes, int n_in,
                              void* d_out, int out_size, void* d_ws, size_t ws_size,
                              hipStream_t stream) {
  (void)in_sizes; (void)n_in; (void)out_size; (void)ws_size;
  const float* x1     = (const float*)d_in[0];
  const float* x2     = (const float*)d_in[1];
  /* d_in[2] clin_var unused by the reference */
  const float* norm_w = (const float*)d_in[3];
  const float* norm_b = (const float*)d_in[4];
  const float* qkvv_W = (const float*)d_in[5];
  const float* E_W    = (const float*)d_in[6];
  const float* E_b    = (const float*)d_in[7];
  const float* F_W    = (const float*)d_in[8];
  const float* F_b    = (const float*)d_in[9];
  const float* opW    = (const float*)d_in[10];
  const float* opB    = (const float*)d_in[11];
  const float* op2W   = (const float*)d_in[12];
  const float* op2B   = (const float*)d_in[13];
  const float* temp   = (const float*)d_in[14];
  const float* temp2  = (const float*)d_in[15];
  const float* gamma1 = (const float*)d_in[16];
  float* out = (float*)d_out;
  float* ws  = (float*)d_ws;

  float* x1f  = ws + O_X1F;
  float* x2f  = ws + O_X2F;
  float* xcat = ws + O_XCAT;
  float* q    = ws + O_Q;
  float* k    = ws + O_K;
  float* vca  = ws + O_VCA;
  float* vsa  = ws + O_VSA;
  float* xca  = ws + O_XCA;
  float* xsa  = ws + O_XSA;
  float* kproj= ws + O_KPROJ;
  float* vproj= ws + O_VPROJ;
  float* qinv = ws + O_QINV;
  float* kinv = ws + O_KINV;
  float* sca  = ws + O_SCA;

  k_layernorm<<<256, 256, 0, stream>>>(x1, norm_w, norm_b, x1f, xcat);
  k_layernorm<<<256, 256, 0, stream>>>(x2, norm_w, norm_b, x2f, xcat + 64);
  k_init<<<64, 256, 0, stream>>>(E_b, F_b, kproj, vproj, sca);
  k_gemm_qkvv<<<16384, 256, 0, stream>>>(xcat, qkvv_W, q, k, vca, vsa);
  k_rowsumsq<<<512, 256, 0, stream>>>(q, k, qinv, kinv);
  // k_proj = k @ E_W^T (+E_b via init): M=32, N=64, split-K 32x1024
  k_gemm_splitk<<<256, 256, 0, stream>>>(k,   (long)DH * NTOK, E_W, 0L,
                                         kproj, (long)(DH * PDIM), PDIM,
                                         2, 4, 32, 1024);
  // v_sa_proj = v_sa @ F_W^T (+F_b)
  k_gemm_splitk<<<256, 256, 0, stream>>>(vsa, (long)DH * NTOK, F_W, 0L,
                                         vproj, (long)(DH * PDIM), PDIM,
                                         2, 4, 32, 1024);
  // raw CA logits q·k^T: M=32, N=32, split-K 64x512
  k_gemm_splitk<<<256, 256, 0, stream>>>(q,   (long)DH * NTOK,
                                         k,   (long)DH * NTOK,
                                         sca, (long)(DH * DH), DH,
                                         2, 2, 64, 512);
  k_softmax_ca<<<256, 32, 0, stream>>>(sca, qinv, kinv, temp);
  k_gemm_xca<<<4096, 256, 0, stream>>>(sca, vca, xca);
  k_sa_fused<<<2048, 256, 0, stream>>>(q, qinv, kproj, vproj, temp2, xsa);
  k_outproj_sa<<<2048, 256, 0, stream>>>(xsa, opW, opB, gamma1, x1f, out);
  k_outproj_ca<<<2048, 256, 0, stream>>>(xca, op2W, op2B, gamma1, x2f,
                                         out + (size_t)BATCH * 64 * NTOK);
}
